// BasicBlock_2662879723628
// MI455X (gfx1250) — 8-bit matrix operand order repaired and checked by reading, compiled, not run on hardware
//
#include <hip/hip_runtime.h>

#define BB 32
#define CC 64
#define HH 112
#define WW 112
#define HWP (HH*WW)
#define ALPHA 0.25f
#define BETA1 1.0f
#define BETA2 1.0f
#define ACT_BYTES ((size_t)BB*HH*WW*CC)
#define WSZ_BYTES (9*4*32*32)

typedef int v8i __attribute__((ext_vector_type(8)));

__device__ __forceinline__ v8i wmma_iu8(v8i a, v8i b, v8i c) {
  return __builtin_amdgcn_wmma_i32_16x16x64_iu8(true, a, true, b, c, false, false);
}

__global__ __launch_bounds__(128) void prep_scales_k(const float* __restrict__ w3,
                                                     const float* __restrict__ wpw,
                                                     float* __restrict__ s1,
                                                     float* __restrict__ s2) {
  const int t = threadIdx.x;
  const float* w = (t < 64) ? w3 : wpw;
  const int co = t & 63;
  float s = 0.f;
  for (int i = 0; i < 576; ++i) s += fabsf(w[co*576 + i]);
  float* dst = (t < 64) ? s1 : s2;
  dst[co] = s * (1.0f/576.0f);
}

__global__ __launch_bounds__(256) void prep_wswz_k(const float* __restrict__ w3,
                                                   const float* __restrict__ wpw,
                                                   signed char* __restrict__ z1,
                                                   signed char* __restrict__ z2) {
  const int t = blockIdx.x*256 + threadIdx.x;
  const int cs = (t >= WSZ_BYTES) ? 1 : 0;
  const int i  = t - cs*WSZ_BYTES;
  const int j    = i & 31;
  const int lane = (i >> 5) & 31;
  const int ct   = (i >> 10) & 3;
  const int tap  = i >> 12;
  const int v = j >> 2, p = j & 3, g = lane >> 4, m = lane & 15;
  const int k  = (v >> 2)*32 + g*16 + (v & 3)*4 + p;
  const int co = ct*16 + m;
  const int kh = tap/3, kw = tap - 3*(tap/3);
  const float* w = cs ? wpw : w3;
  const signed char s = (w[((co*64 + k)*3 + kh)*3 + kw] >= 0.f) ? (signed char)1
                                                                : (signed char)-1;
  (cs ? z2 : z1)[i] = s;
}

__global__ __launch_bounds__(256) void binarize_x_k(const float* __restrict__ x,
                                                    const float* __restrict__ b11,
                                                    signed char* __restrict__ act1) {
  const int pid = blockIdx.x*256 + threadIdx.x;
  const int b = pid / HWP, rem = pid - b*HWP;
  const float* xb = x + (size_t)b*CC*HWP + rem;
  unsigned int pk[16];
#pragma unroll
  for (int cq = 0; cq < 16; ++cq) {
    unsigned int word = 0;
#pragma unroll
    for (int s = 0; s < 4; ++s) {
      const int c = cq*4 + s;
      const float v = xb[(size_t)c*HWP]*BETA1 + b11[c];
      word |= (unsigned int)((v >= 0.f) ? 0x01u : 0xFFu) << (8*s);
    }
    pk[cq] = word;
  }
  uint4* dst = (uint4*)(act1 + (size_t)pid*64);
  uint4 q;
  q.x = pk[0];  q.y = pk[1];  q.z = pk[2];  q.w = pk[3];  dst[0] = q;
  q.x = pk[4];  q.y = pk[5];  q.z = pk[6];  q.w = pk[7];  dst[1] = q;
  q.x = pk[8];  q.y = pk[9];  q.z = pk[10]; q.w = pk[11]; dst[2] = q;
  q.x = pk[12]; q.y = pk[13]; q.z = pk[14]; q.w = pk[15]; dst[3] = q;
}

__device__ __forceinline__ v8i load_bmat(const signed char* rowp, int xx, int g) {
  const bool valid = (xx >= 0) & (xx < WW);
  const int xc = valid ? xx : 0;
  const int4* p = (const int4*)(rowp + (size_t)xc*64 + g*16);
  const int4 lo = p[0];
  const int4 hi = p[2];
  v8i bm;
  bm[0] = valid ? lo.x : 0; bm[1] = valid ? lo.y : 0;
  bm[2] = valid ? lo.z : 0; bm[3] = valid ? lo.w : 0;
  bm[4] = valid ? hi.x : 0; bm[5] = valid ? hi.y : 0;
  bm[6] = valid ? hi.z : 0; bm[7] = valid ? hi.w : 0;
  return bm;
}

__device__ __forceinline__ void epi1(v8i acc, int x0, int n, int b, int y,
                                     int co_base,
                                     const float* sc, const float* c12,
                                     const float* c13, const float* ca,
                                     const float* c21,
                                     const float* __restrict__ x,
                                     float* __restrict__ out1,
                                     signed char* __restrict__ act2) {
  const int px = x0 + n;
  unsigned long long pk = 0;
#pragma unroll
  for (int r = 0; r < 8; ++r) {
    const size_t idx = ((size_t)(b*CC + co_base + r)*HH + y)*WW + px;
    float o1 = x[idx] + (float)acc[r]*sc[r] + c12[r];
    o1 = (o1 >= 0.f) ? o1 : ca[r]*o1;
    o1 += c13[r];
    out1[idx] = o1;
    const unsigned long long sb = ((o1*BETA2 + c21[r]) >= 0.f) ? 0x01ull : 0xFFull;
    pk |= sb << (8*r);
  }
  *(unsigned long long*)(act2 + (size_t)((b*HH + y)*WW + px)*64 + co_base) = pk;
}

__device__ __forceinline__ void epi2(v8i acc, int x0, int n, int b, int y,
                                     int co_base,
                                     const float* sc, const float* c22,
                                     const float* c23, const float* ca,
                                     float* io) {
  const int px = x0 + n;
#pragma unroll
  for (int r = 0; r < 8; ++r) {
    const size_t idx = ((size_t)(b*CC + co_base + r)*HH + y)*WW + px;
    float o2 = (float)acc[r]*sc[r] + io[idx] + c22[r];
    o2 = (o2 >= 0.f) ? o2 : ca[r]*o2;
    o2 += c23[r];
    io[idx] = o2;
  }
}

__global__ __launch_bounds__(128) void conv1_fused_k(
    const signed char* __restrict__ act1, const int* __restrict__ wsz,
    const float* __restrict__ x, const float* __restrict__ scale,
    const float* __restrict__ b12, const float* __restrict__ b13,
    const float* __restrict__ a1, const float* __restrict__ b21,
    float* __restrict__ out1, signed char* __restrict__ act2) {
  __shared__ int ldsw[WSZ_BYTES/4];
  const int tid = threadIdx.x;
#pragma unroll 4
  for (int i = tid; i < WSZ_BYTES/4; i += 128) ldsw[i] = wsz[i];
  __syncthreads();

  const int lane = tid & 31, wv = tid >> 5;
  const int g = lane >> 4, n = lane & 15;
  const int bh = blockIdx.x, b = bh / HH, y = bh - b*HH;
  const int co_base = wv*16 + g*8;

  float sc[8], c12[8], c13[8], ca[8], c21[8];
#pragma unroll
  for (int r = 0; r < 8; ++r) {
    sc[r]  = scale[co_base + r] * ALPHA;
    c12[r] = b12[co_base + r];
    c13[r] = b13[co_base + r];
    ca[r]  = a1[co_base + r];
    c21[r] = b21[co_base + r];
  }

  for (int xp = 0; xp < 3; ++xp) {
    const int x0 = xp*32;
    v8i acc0 = {0,0,0,0,0,0,0,0};
    v8i acc1 = {0,0,0,0,0,0,0,0};
#pragma unroll
    for (int kh = 0; kh < 3; ++kh) {
      const int yy = y + kh - 1;
      if (yy < 0 || yy >= HH) continue;
      const signed char* rowp = act1 + (size_t)((b*HH + yy)*WW)*64;
      __builtin_prefetch(rowp + (size_t)(x0 + 32)*64 + (size_t)n*64, 0, 0);
#pragma unroll
      for (int kw = 0; kw < 3; ++kw) {
        const v8i bm0 = load_bmat(rowp, x0 + n + kw - 1, g);
        const v8i bm1 = load_bmat(rowp, x0 + 16 + n + kw - 1, g);
        const v8i am = *(const v8i*)&ldsw[(((kh*3 + kw)*4 + wv)*32 + lane)*8];
        acc0 = wmma_iu8(am, bm0, acc0);
        acc1 = wmma_iu8(am, bm1, acc1);
      }
    }
    epi1(acc0, x0,      n, b, y, co_base, sc, c12, c13, ca, c21, x, out1, act2);
    epi1(acc1, x0 + 16, n, b, y, co_base, sc, c12, c13, ca, c21, x, out1, act2);
  }
  {
    v8i acc = {0,0,0,0,0,0,0,0};
#pragma unroll
    for (int kh = 0; kh < 3; ++kh) {
      const int yy = y + kh - 1;
      if (yy < 0 || yy >= HH) continue;
      const signed char* rowp = act1 + (size_t)((b*HH + yy)*WW)*64;
#pragma unroll
      for (int kw = 0; kw < 3; ++kw) {
        const v8i bm = load_bmat(rowp, 96 + n + kw - 1, g);
        const v8i am = *(const v8i*)&ldsw[(((kh*3 + kw)*4 + wv)*32 + lane)*8];
        acc = wmma_iu8(am, bm, acc);
      }
    }
    epi1(acc, 96, n, b, y, co_base, sc, c12, c13, ca, c21, x, out1, act2);
  }
}

__global__ __launch_bounds__(128) void conv2_fused_k(
    const signed char* __restrict__ act2, const int* __restrict__ wsz,
    const float* __restrict__ scale, const float* __restrict__ b22,
    const float* __restrict__ b23, const float* __restrict__ a2,
    float* io) {
  __shared__ int ldsw[WSZ_BYTES/4];
  const int tid = threadIdx.x;
#pragma unroll 4
  for (int i = tid; i < WSZ_BYTES/4; i += 128) ldsw[i] = wsz[i];
  __syncthreads();

  const int lane = tid & 31, wv = tid >> 5;
  const int g = lane >> 4, n = lane & 15;
  const int bh = blockIdx.x, b = bh / HH, y = bh - b*HH;
  const int co_base = wv*16 + g*8;

  float sc[8], c22[8], c23[8], ca[8];
#pragma unroll
  for (int r = 0; r < 8; ++r) {
    sc[r]  = scale[co_base + r] * ALPHA;
    c22[r] = b22[co_base + r];
    c23[r] = b23[co_base + r];
    ca[r]  = a2[co_base + r];
  }

  for (int xp = 0; xp < 3; ++xp) {
    const int x0 = xp*32;
    v8i acc0 = {0,0,0,0,0,0,0,0};
    v8i acc1 = {0,0,0,0,0,0,0,0};
#pragma unroll
    for (int kh = 0; kh < 3; ++kh) {
      const int yy = y + kh - 1;
      if (yy < 0 || yy >= HH) continue;
      const signed char* rowp = act2 + (size_t)((b*HH + yy)*WW)*64;
      __builtin_prefetch(rowp + (size_t)(x0 + 32)*64 + (size_t)n*64, 0, 0);
#pragma unroll
      for (int kw = 0; kw < 3; ++kw) {
        const v8i bm0 = load_bmat(rowp, x0 + n + kw - 1, g);
        const v8i bm1 = load_bmat(rowp, x0 + 16 + n + kw - 1, g);
        const v8i am = *(const v8i*)&ldsw[(((kh*3 + kw)*4 + wv)*32 + lane)*8];
        acc0 = wmma_iu8(am, bm0, acc0);
        acc1 = wmma_iu8(am, bm1, acc1);
      }
    }
    epi2(acc0, x0,      n, b, y, co_base, sc, c22, c23, ca, io);
    epi2(acc1, x0 + 16, n, b, y, co_base, sc, c22, c23, ca, io);
  }
  {
    v8i acc = {0,0,0,0,0,0,0,0};
#pragma unroll
    for (int kh = 0; kh < 3; ++kh) {
      const int yy = y + kh - 1;
      if (yy < 0 || yy >= HH) continue;
      const signed char* rowp = act2 + (size_t)((b*HH + yy)*WW)*64;
#pragma unroll
      for (int kw = 0; kw < 3; ++kw) {
        const v8i bm = load_bmat(rowp, 96 + n + kw - 1, g);
        const v8i am = *(const v8i*)&ldsw[(((kh*3 + kw)*4 + wv)*32 + lane)*8];
        acc = wmma_iu8(am, bm, acc);
      }
    }
    epi2(acc, 96, n, b, y, co_base, sc, c22, c23, ca, io);
  }
}

extern "C" void kernel_launch(void* const* d_in, const int* in_sizes, int n_in,
                              void* d_out, int out_size, void* d_ws, size_t ws_size,
                              hipStream_t stream) {
  (void)in_sizes; (void)n_in; (void)out_size; (void)ws_size;
  const float* x   = (const float*)d_in[0];
  const float* w3  = (const float*)d_in[1];
  const float* wpw = (const float*)d_in[2];
  const float* b11 = (const float*)d_in[3];
  const float* b12 = (const float*)d_in[4];
  const float* b13 = (const float*)d_in[5];
  const float* b21 = (const float*)d_in[6];
  const float* b22 = (const float*)d_in[7];
  const float* b23 = (const float*)d_in[8];
  const float* a1  = (const float*)d_in[9];
  const float* a2  = (const float*)d_in[10];
  float* out = (float*)d_out;

  char* ws = (char*)d_ws;
  size_t o = 0;
  signed char* act1 = (signed char*)(ws + o); o += ACT_BYTES;
  signed char* act2 = (signed char*)(ws + o); o += ACT_BYTES;
  signed char* wsz1 = (signed char*)(ws + o); o += WSZ_BYTES;
  signed char* wsz2 = (signed char*)(ws + o); o += WSZ_BYTES;
  float* s1 = (float*)(ws + o); o += 256;
  float* s2 = (float*)(ws + o); o += 256;

  prep_scales_k<<<1, 128, 0, stream>>>(w3, wpw, s1, s2);
  prep_wswz_k<<<(2*WSZ_BYTES)/256, 256, 0, stream>>>(w3, wpw, wsz1, wsz2);
  binarize_x_k<<<(BB*HWP)/256, 256, 0, stream>>>(x, b11, act1);
  conv1_fused_k<<<BB*HH, 128, 0, stream>>>(act1, (const int*)wsz1, x, s1,
                                           b12, b13, a1, b21, out, act2);
  conv2_fused_k<<<BB*HH, 128, 0, stream>>>(act2, (const int*)wsz2, s2,
                                           b22, b23, a2, out);
}
